// ATTN_37641093382453
// MI455X (gfx1250) — compile-verified, matrix operand order checked by reading, not run on hardware
//
#include <hip/hip_runtime.h>
#include <hip/hip_bf16.h>

// ---------------------------------------------------------------------------
// Temporal graph attention, 2 layers, for MI455X (gfx1250, wave32, WMMA).
// Heavy GEMMs run on v_wmma_f32_16x16x32_f16 with 32x64 register blocking
// (8 accumulators/wave -> 1.5 b128 loads per WMMA). Intermediates chunked
// (2048 nodes -> ~60MB working set) to stay resident in the 192MB L2.
// ---------------------------------------------------------------------------

#define NN   16384
#define KK   16
#define LL   2
#define NFD  128
#define TD   128
#define EFD  128
#define QDIM 256            // NFD + TD
#define KDIM 384            // NFD + TD + EFD
#define HH   4
#define HD   64
#define EMB  128
#define XDIM (NFD + QDIM)   // 384 merge input
#define CH_NODES 2048
#define CH_ROWS  (CH_NODES * KK)   // 32768
#define NCHUNK   (NN / CH_NODES)   // 8

typedef __attribute__((ext_vector_type(16))) _Float16 v16h;
typedef __attribute__((ext_vector_type(8)))  float    v8f;

union Frag16 { v16h h; int4 q[2]; };

// ---------------------------------------------------------------------------
// Generic f16 GEMM:  out[M,Ncols] = A[M,Kdim] * W[Ncols,Kdim]^T + bias
// One wave computes a 32x64 block = 2 M-tiles x 4 N-tiles, K stepped by 32.
// Fragment loads follow the CDNA5 16-bit A/B layout: lane = row (A) / col (B),
// halves 0..7 = K (lane/16)*8 + 0..7, halves 8..15 = same + 16 -> two b128
// loads per fragment from row-major memory. Per K-step: 12 loads, 8 WMMAs.
// ---------------------------------------------------------------------------
__global__ __launch_bounds__(256) void gemm_f16_wmma(
    const _Float16* __restrict__ A, int lda,
    const _Float16* __restrict__ W, int ldw,
    const float* __restrict__ bias,
    int M, int Kdim, int Ncols,
    _Float16* __restrict__ outH, float* __restrict__ outF,
    int ldo, int ocol, int relu)
{
    const int wave = threadIdx.x >> 5;
    const int lane = threadIdx.x & 31;
    const int lr   = lane & 15;
    const int hi8  = (lane >> 4) * 8;

    const int MB = M >> 5;                     // # of 32-row blocks
    const int t  = blockIdx.x * 8 + wave;
    const int mb = t % MB;                     // consecutive waves -> consecutive M
    const int nb = t / MB;                     // same N block -> B frag reuse in L0
    const int m0 = mb << 5;
    const int n0 = nb << 6;

    const _Float16* ap0 = A + (size_t)(m0 + lr) * lda + hi8;
    const _Float16* ap1 = ap0 + (size_t)16 * lda;
    const _Float16* bp  = W + (size_t)(n0 + lr) * ldw + hi8;
    const size_t bstep  = (size_t)16 * ldw;    // next 16 output columns

    v8f acc[2][4];
#pragma unroll
    for (int i = 0; i < 2; ++i)
#pragma unroll
        for (int j = 0; j < 4; ++j)
            acc[i][j] = (v8f){0.f, 0.f, 0.f, 0.f, 0.f, 0.f, 0.f, 0.f};

    for (int k0 = 0; k0 < Kdim; k0 += 32) {
        Frag16 a0, a1, b[4];
        a0.q[0] = *reinterpret_cast<const int4*>(ap0 + k0);
        a0.q[1] = *reinterpret_cast<const int4*>(ap0 + k0 + 16);
        a1.q[0] = *reinterpret_cast<const int4*>(ap1 + k0);
        a1.q[1] = *reinterpret_cast<const int4*>(ap1 + k0 + 16);
#pragma unroll
        for (int j = 0; j < 4; ++j) {
            b[j].q[0] = *reinterpret_cast<const int4*>(bp + j * bstep + k0);
            b[j].q[1] = *reinterpret_cast<const int4*>(bp + j * bstep + k0 + 16);
        }
#pragma unroll
        for (int j = 0; j < 4; ++j) {
            acc[0][j] = __builtin_amdgcn_wmma_f32_16x16x32_f16(
                false, a0.h, false, b[j].h, (short)0, acc[0][j], false, false);
            acc[1][j] = __builtin_amdgcn_wmma_f32_16x16x32_f16(
                false, a1.h, false, b[j].h, (short)0, acc[1][j], false, false);
        }
    }

    // C/D layout: lane<16 -> (M=r, N=lane); lane>=16 -> (M=8+r, N=lane-16)
#pragma unroll
    for (int j = 0; j < 4; ++j) {
        const int   col = n0 + j * 16 + lr;
        const float bc  = bias ? bias[col] : 0.f;
#pragma unroll
        for (int i = 0; i < 2; ++i) {
            const int rbase = m0 + i * 16 + hi8;
#pragma unroll
            for (int r = 0; r < 8; ++r) {
                float v = acc[i][j][r] + bc;
                if (relu) v = fmaxf(v, 0.f);
                const size_t o = (size_t)(rbase + r) * ldo + ocol + col;
                if (outH) outH[o] = (_Float16)v;
                else      outF[o] = v;
            }
        }
    }
}

// ---------------------------------------------------------------------------
// kv[r, :] = [ h[idx] | edge_feat | cos(dt*time_w + time_b) ]  (f16)
// grid: (CH_ROWS, 3), block: 128
// ---------------------------------------------------------------------------
__global__ void build_kv_kernel(
    const float* __restrict__ h,
    const int*   __restrict__ nbr,     // layer slice [N*K]
    const float* __restrict__ ef,      // layer slice [N*K*EFD]
    const float* __restrict__ ts,      // layer slice [N*K]
    const float* __restrict__ lu,      // [N]
    const float* __restrict__ tw,
    const float* __restrict__ tb,
    _Float16* __restrict__ kv, int row0)
{
    const int r   = blockIdx.x;
    const int c   = blockIdx.y * 128 + threadIdx.x;
    const int rg  = row0 + r;
    const int idx = nbr[rg];
    float v;
    if (c < NFD) {
        v = h[(size_t)idx * NFD + c];
    } else if (c < NFD + EFD) {
        v = ef[(size_t)rg * EFD + (c - NFD)];
    } else {
        const int d = c - (NFD + EFD);
        const float dt = ts[rg] - lu[idx];
        v = cosf(dt * tw[d] + tb[d]);
    }
    kv[(size_t)r * KDIM + c] = (_Float16)v;
}

// query[n, :] = [ h[n] | cos(time_b) ]   grid: NN, block: 256
__global__ void build_query_kernel(const float* __restrict__ h,
                                   const float* __restrict__ tb,
                                   _Float16* __restrict__ q)
{
    const int n = blockIdx.x;
    const int c = threadIdx.x;
    const float v = (c < NFD) ? h[(size_t)n * NFD + c] : cosf(tb[c - NFD]);
    q[(size_t)n * QDIM + c] = (_Float16)v;
}

// x[n, 0:128] = f16(h[n])   (h_before goes to cols 128..383 via Wo GEMM ocol)
__global__ void copy_h_kernel(const float* __restrict__ h, _Float16* __restrict__ x)
{
    const int n = blockIdx.x;
    const int c = threadIdx.x;
    x[(size_t)n * XDIM + c] = (_Float16)h[(size_t)n * NFD + c];
}

__global__ void f32_to_f16_kernel(const float* __restrict__ in,
                                  _Float16* __restrict__ out, int n)
{
    const int i = blockIdx.x * 256 + threadIdx.x;
    if (i < n) out[i] = (_Float16)in[i];
}

// ---------------------------------------------------------------------------
// Fused scores / softmax / context for one chunk. Block = 4 nodes x 64 thr.
// Thread s = h*16+j computes score[h][j]; width-16 shfl reductions (wave32).
// ---------------------------------------------------------------------------
__global__ __launch_bounds__(256) void attn_kernel(
    const _Float16* __restrict__ q,    // [N, QDIM]
    const _Float16* __restrict__ Kp,   // [CH_ROWS, QDIM] chunk-local
    const _Float16* __restrict__ Vp,   // [CH_ROWS, QDIM] chunk-local
    _Float16* __restrict__ ctx,        // [N, QDIM]
    float* __restrict__ att_avg,       // [N*K]
    float* __restrict__ att_sum,       // scalar accumulator
    int node0, int accum)
{
    __shared__ float sAtt[4][HH][KK];
    const int g    = threadIdx.x >> 6;
    const int s    = threadIdx.x & 63;
    const int hId  = s >> 4;
    const int j    = s & 15;
    const int nloc = blockIdx.x * 4 + g;
    const int n    = node0 + nloc;
    const int rl   = nloc * KK + j;

    const _Float16* qb = q  + (size_t)n  * QDIM + hId * HD;
    const _Float16* kb = Kp + (size_t)rl * QDIM + hId * HD;
    float sc = 0.f;
#pragma unroll
    for (int d = 0; d < HD; ++d) sc += (float)qb[d] * (float)kb[d];
    sc *= 0.125f;  // 1/sqrt(64)

    float mx = sc;
    for (int o = 8; o >= 1; o >>= 1) mx = fmaxf(mx, __shfl_xor(mx, o, 16));
    const float e = __expf(sc - mx);
    float sm = e;
    for (int o = 8; o >= 1; o >>= 1) sm += __shfl_xor(sm, o, 16);
    const float att = e / sm;
    sAtt[g][hId][j] = att;
    __syncthreads();

    if (s < KK) {
        const float avg = 0.25f * (sAtt[g][0][s] + sAtt[g][1][s] +
                                   sAtt[g][2][s] + sAtt[g][3][s]);
        att_avg[(size_t)n * KK + s] = avg;
        if (accum) {
            float t = avg;
            for (int o = 8; o >= 1; o >>= 1) t += __shfl_xor(t, o, 16);
            if (s == 0) atomicAdd(att_sum, t);
        }
    }

    const _Float16* vb = Vp + (size_t)(nloc * KK) * QDIM;
#pragma unroll
    for (int hb = 0; hb < HH; ++hb) {
        const int qd = hb * HD + s;     // s in [0,64) == HD, so head(qd) == hb
        float acc = 0.f;
#pragma unroll
        for (int jj = 0; jj < KK; ++jj)
            acc += sAtt[g][hb][jj] * (float)vb[(size_t)jj * QDIM + qd];
        ctx[(size_t)n * QDIM + qd] = (_Float16)acc;
    }
}

__global__ void zero2_kernel(float* p) { if (threadIdx.x < 2) p[threadIdx.x] = 0.f; }

__global__ void reduce_sum_kernel(const float* __restrict__ x, int n,
                                  float* __restrict__ out)
{
    __shared__ float sd[256];
    float s = 0.f;
    for (int i = blockIdx.x * 256 + threadIdx.x; i < n; i += gridDim.x * 256)
        s += x[i];
    sd[threadIdx.x] = s; __syncthreads();
    for (int o = 128; o > 0; o >>= 1) {
        if ((int)threadIdx.x < o) sd[threadIdx.x] += sd[threadIdx.x + o];
        __syncthreads();
    }
    if (threadIdx.x == 0) atomicAdd(out, sd[0]);
}

// attn_map[i] = qij_sum * att_avg[i] / att_sum
__global__ void finalize_kernel(const float* __restrict__ att_avg,
                                const float* __restrict__ scal,
                                float* __restrict__ out)
{
    const int i = blockIdx.x * 256 + threadIdx.x;
    out[i] = scal[0] * att_avg[i] / scal[1];
}

// ---------------------------------------------------------------------------
extern "C" void kernel_launch(void* const* d_in, const int* in_sizes, int n_in,
                              void* d_out, int out_size, void* d_ws, size_t ws_size,
                              hipStream_t stream)
{
    (void)in_sizes; (void)n_in; (void)out_size; (void)ws_size;

    const float* h0  = (const float*)d_in[0];
    const int*   nbr = (const int*)  d_in[1];
    const float* ef  = (const float*)d_in[2];
    const float* ts  = (const float*)d_in[3];
    const float* lu  = (const float*)d_in[4];
    const float* qij = (const float*)d_in[5];
    const float* tw  = (const float*)d_in[6];
    const float* tb  = (const float*)d_in[7];
    const float* Wq  = (const float*)d_in[8];
    const float* Wk  = (const float*)d_in[9];
    const float* Wv  = (const float*)d_in[10];
    const float* bq  = (const float*)d_in[11];
    const float* bk  = (const float*)d_in[12];
    const float* bv  = (const float*)d_in[13];
    const float* Wo  = (const float*)d_in[14];
    const float* bo  = (const float*)d_in[15];
    const float* f1w = (const float*)d_in[16];
    const float* f1b = (const float*)d_in[17];
    const float* f2w = (const float*)d_in[18];
    const float* f2b = (const float*)d_in[19];
    float* out = (float*)d_out;

    // ---- workspace layout -------------------------------------------------
    char*  ws  = (char*)d_ws;
    size_t off = 0;
    auto alloc = [&](size_t bytes) -> void* {
        void* p = ws + off;
        off = (off + bytes + 255) & ~(size_t)255;
        return p;
    };
    _Float16* Wq_h  = (_Float16*)alloc((size_t)LL * QDIM * QDIM * 2);
    _Float16* Wk_h  = (_Float16*)alloc((size_t)LL * QDIM * KDIM * 2);
    _Float16* Wv_h  = (_Float16*)alloc((size_t)LL * QDIM * KDIM * 2);
    _Float16* Wo_h  = (_Float16*)alloc((size_t)LL * QDIM * QDIM * 2);
    _Float16* f1w_h = (_Float16*)alloc((size_t)LL * NFD * XDIM * 2);
    _Float16* f2w_h = (_Float16*)alloc((size_t)LL * EMB * NFD * 2);
    float*    hbuf  = (float*)   alloc((size_t)NN * NFD * 4);
    _Float16* qry_h = (_Float16*)alloc((size_t)NN * QDIM * 2);
    _Float16* q_h   = (_Float16*)alloc((size_t)NN * QDIM * 2);
    _Float16* kv_h  = (_Float16*)alloc((size_t)CH_ROWS * KDIM * 2);
    _Float16* Kp_h  = (_Float16*)alloc((size_t)CH_ROWS * QDIM * 2);
    _Float16* Vp_h  = (_Float16*)alloc((size_t)CH_ROWS * QDIM * 2);
    _Float16* ctx_h = (_Float16*)alloc((size_t)NN * QDIM * 2);
    _Float16* x_h   = (_Float16*)alloc((size_t)NN * XDIM * 2);
    _Float16* fo1_h = (_Float16*)alloc((size_t)NN * NFD * 2);
    float*    aavg  = (float*)   alloc((size_t)NN * KK * 4);
    float*    scal  = (float*)   alloc(2 * 4);   // [0]=qij_sum [1]=att_sum

    // ---- one-time (per call) weight conversion + scalar zeroing -----------
    auto cvt = [&](const float* src, _Float16* dst, int n) {
        f32_to_f16_kernel<<<(n + 255) / 256, 256, 0, stream>>>(src, dst, n);
    };
    cvt(Wq,  Wq_h,  LL * QDIM * QDIM);
    cvt(Wk,  Wk_h,  LL * QDIM * KDIM);
    cvt(Wv,  Wv_h,  LL * QDIM * KDIM);
    cvt(Wo,  Wo_h,  LL * QDIM * QDIM);
    cvt(f1w, f1w_h, LL * NFD * XDIM);
    cvt(f2w, f2w_h, LL * EMB * NFD);
    zero2_kernel<<<1, 32, 0, stream>>>(scal);

    // GEMM grid helper: 32x64 block per wave, 8 waves per workgroup
    auto gemm_blocks = [](int M, int Ncols) {
        return (M / 32) * (Ncols / 64) / 8;
    };

    for (int l = 0; l < LL; ++l) {
        const float* hcur = (l == 0) ? h0 : hbuf;

        // query = [h | cos(time_b)] ; q = query @ Wq^T + bq
        build_query_kernel<<<NN, 256, 0, stream>>>(hcur, tb, qry_h);
        gemm_f16_wmma<<<gemm_blocks(NN, QDIM), 256, 0, stream>>>(
            qry_h, QDIM, Wq_h + (size_t)l * QDIM * QDIM, QDIM, bq + l * QDIM,
            NN, QDIM, QDIM, q_h, nullptr, QDIM, 0, 0);

        for (int c = 0; c < NCHUNK; ++c) {
            const int row0 = c * CH_ROWS;
            build_kv_kernel<<<dim3(CH_ROWS, 3), 128, 0, stream>>>(
                hcur, nbr + (size_t)l * NN * KK, ef + (size_t)l * NN * KK * EFD,
                ts + (size_t)l * NN * KK, lu, tw, tb, kv_h, row0);
            gemm_f16_wmma<<<gemm_blocks(CH_ROWS, QDIM), 256, 0, stream>>>(
                kv_h, KDIM, Wk_h + (size_t)l * QDIM * KDIM, KDIM, bk + l * QDIM,
                CH_ROWS, KDIM, QDIM, Kp_h, nullptr, QDIM, 0, 0);
            gemm_f16_wmma<<<gemm_blocks(CH_ROWS, QDIM), 256, 0, stream>>>(
                kv_h, KDIM, Wv_h + (size_t)l * QDIM * KDIM, KDIM, bv + l * QDIM,
                CH_ROWS, KDIM, QDIM, Vp_h, nullptr, QDIM, 0, 0);
            attn_kernel<<<CH_NODES / 4, 256, 0, stream>>>(
                q_h, Kp_h, Vp_h, ctx_h, aavg, scal + 1,
                c * CH_NODES, (l == LL - 1) ? 1 : 0);
        }

        // h_before = ctx @ Wo^T + bo  -> x[:,128:384] ; x[:,0:128] = h
        gemm_f16_wmma<<<gemm_blocks(NN, QDIM), 256, 0, stream>>>(
            ctx_h, QDIM, Wo_h + (size_t)l * QDIM * QDIM, QDIM, bo + l * QDIM,
            NN, QDIM, QDIM, x_h, nullptr, XDIM, NFD, 0);
        copy_h_kernel<<<NN, 128, 0, stream>>>(hcur, x_h);

        // h = fc2( relu( fc1(x) ) )
        gemm_f16_wmma<<<gemm_blocks(NN, NFD), 256, 0, stream>>>(
            x_h, XDIM, f1w_h + (size_t)l * NFD * XDIM, XDIM, f1b + l * NFD,
            NN, XDIM, NFD, fo1_h, nullptr, NFD, 0, 1);
        gemm_f16_wmma<<<gemm_blocks(NN, EMB), 256, 0, stream>>>(
            fo1_h, NFD, f2w_h + (size_t)l * EMB * NFD, NFD, f2b + l * EMB,
            NN, NFD, EMB, nullptr, (l == 0) ? hbuf : out, EMB, 0, 0);
    }

    // attn_map = qij_sum * att_avg / att_sum   (layer L-1)
    reduce_sum_kernel<<<256, 256, 0, stream>>>(
        qij + (size_t)(LL - 1) * NN * KK, NN * KK, scal);
    finalize_kernel<<<(NN * KK) / 256, 256, 0, stream>>>(
        aavg, scal, out + (size_t)NN * EMB);
}
